// WindowAttention_33964601377146
// MI455X (gfx1250) — compile-verified, matrix operand order checked by reading, not run on hardware
//
#include <hip/hip_runtime.h>
#include <hip/hip_bf16.h>

typedef __bf16 bf16;
typedef __attribute__((ext_vector_type(16))) __bf16 v16bf;
typedef __attribute__((ext_vector_type(8)))  float  v8f;

#define B_TOT 512
#define SEQ_N 145
#define CDIM  512
#define HEADS 8
#define HD    64
#define NWIN  4
#define MTOK  144

__device__ __forceinline__ v8f wmma_bf16(v16bf a, v16bf b, v8f c) {
  // D = A(16x32 bf16) x B(32x16 bf16) + C(16x16 f32)
  return __builtin_amdgcn_wmma_f32_16x16x32_bf16(false, a, false, b, (short)0, c, false, false);
}

// Async copy 16B global -> LDS, tracked on ASYNCcnt (CDNA5 TDM-lite path).
// VDST = VGPR holding LDS byte address, VADDR = 64-bit global address.
__device__ __forceinline__ void async_ld_b128(void* lds, const void* gptr) {
  unsigned l = (unsigned)(unsigned long long)lds;   // low 32 bits of flat ptr = LDS offset
  asm volatile("global_load_async_to_lds_b128 %0, %1, off"
               :: "v"(l), "v"(gptr) : "memory");
}
__device__ __forceinline__ void wait_async_le4() {
  asm volatile("s_wait_asynccnt 4" ::: "memory");
}
__device__ __forceinline__ void wait_async_0() {
  asm volatile("s_wait_asynccnt 0" ::: "memory");
}

union FragU { v16bf v; unsigned int u[8]; bf16 h[16]; };

// Gather a 16x32 bf16 fragment from a row-major [row][K] LDS tile.
// lane<16 -> row=lane, K in {0..7,16..23}; lane>=16 -> row=lane-16, K in {8..15,24..31}.
__device__ __forceinline__ v16bf load_frag_lds_bf16(const bf16* base, int row0, int k0, int ld, int lane) {
  int r = lane & 15, hf = lane >> 4;
  const bf16* p = base + (size_t)(row0 + r) * ld + k0 + hf * 8;
  FragU f;
#pragma unroll
  for (int i = 0; i < 8; ++i) {
    int K = 2 * i + ((i >= 4) ? 8 : 0);
    f.u[i] = *(const unsigned int*)(p + K);   // pair of bf16, 4B-aligned
  }
  return f.v;
}

// Same gather but from an f32 LDS tile, converting to bf16 on the fly.
__device__ __forceinline__ v16bf load_frag_lds_f32(const float* base, int row0, int k0, int ld, int lane) {
  int r = lane & 15, hf = lane >> 4;
  const float* p = base + (size_t)(row0 + r) * ld + k0 + hf * 8;
  FragU f;
#pragma unroll
  for (int e = 0; e < 16; ++e) {
    int K = e + ((e >= 8) ? 8 : 0);
    f.h[e] = (bf16)p[K];
  }
  return f.v;
}

// ---------------------------------------------------------------- converts
__global__ __launch_bounds__(256) void cvt_f32_to_bf16(const float* __restrict__ in,
                                                       bf16* __restrict__ out, int n) {
  int i = blockIdx.x * 256 + threadIdx.x;
  if (i < n) out[i] = (bf16)in[i];
}

// ---------------------------------------------------------------- QKV GEMM
// out[m, c] = sum_k x[m,k] * W[c,k] + bias[c]; m = b_*145+n, c = h*64+d
// written bf16 as [B_, H, N, hd]. 64x64 block tile, K-stage 64, double-buffered
// LDS fed by global_load_async_to_lds_b128 (ASYNCcnt), 8 waves x 2 WMMA tiles.
__global__ __launch_bounds__(256) void qkv_gemm_kernel(const bf16* __restrict__ xb,
                                                       const bf16* __restrict__ wb,
                                                       const float* __restrict__ bq,
                                                       const float* __restrict__ bk,
                                                       const float* __restrict__ bv,
                                                       bf16* __restrict__ q,
                                                       bf16* __restrict__ k,
                                                       bf16* __restrict__ v) {
  __shared__ __align__(16) bf16 As[2][64][72];
  __shared__ __align__(16) bf16 Bs[2][64][72];
  int z = blockIdx.z;
  const bf16* w = wb + (size_t)z * CDIM * CDIM;
  const float* bias = (z == 0) ? bq : (z == 1) ? bk : bv;
  bf16* outp = (z == 0) ? q : (z == 1) ? k : v;
  int mtile = blockIdx.x, ntile = blockIdx.y;
  int tid = threadIdx.x, lane = tid & 31, wv = tid >> 5;
  int hf = lane >> 4, ln = lane & 15;
  int m0 = mtile * 64, n0 = ntile * 64;

  auto issue_stage = [&](int buf, int kc) {
#pragma unroll
    for (int c = tid; c < 512; c += 256) {
      int r = c >> 3, kk = (c & 7) * 8;
      async_ld_b128(&As[buf][r][kk], xb + (size_t)(m0 + r) * CDIM + kc + kk);
      async_ld_b128(&Bs[buf][r][kk], w  + (size_t)(n0 + r) * CDIM + kc + kk);
    }
  };

  v8f acc0 = {}; v8f acc1 = {};
  int buf = 0;
  issue_stage(0, 0);
  for (int kc = 0; kc < CDIM; kc += 64) {
    if (kc + 64 < CDIM) { issue_stage(buf ^ 1, kc + 64); wait_async_le4(); }
    else                { wait_async_0(); }
    __syncthreads();
#pragma unroll
    for (int k0 = 0; k0 < 64; k0 += 32) {
      v16bf bf = load_frag_lds_bf16(&Bs[buf][0][0], (wv & 3) * 16, k0, 72, lane);
      v16bf a0 = load_frag_lds_bf16(&As[buf][0][0], (wv >> 2) * 16, k0, 72, lane);
      v16bf a1 = load_frag_lds_bf16(&As[buf][0][0], ((wv >> 2) + 2) * 16, k0, 72, lane);
      acc0 = wmma_bf16(a0, bf, acc0);
      acc1 = wmma_bf16(a1, bf, acc1);
    }
    __syncthreads();
    buf ^= 1;
  }
  int c = n0 + (wv & 3) * 16 + ln;
  float bc = bias[c];
  int hh = c >> 6, dd = c & 63;
#pragma unroll
  for (int i = 0; i < 8; ++i) {
    int r0 = m0 + (wv >> 2) * 16 + i + hf * 8;
    int b0 = r0 / SEQ_N, q0 = r0 % SEQ_N;
    outp[((size_t)(b0 * HEADS + hh) * SEQ_N + q0) * HD + dd] = (bf16)(acc0[i] + bc);
    int r1 = r0 + 32;
    int b1 = r1 / SEQ_N, q1 = r1 % SEQ_N;
    outp[((size_t)(b1 * HEADS + hh) * SEQ_N + q1) * HD + dd] = (bf16)(acc1[i] + bc);
  }
}

// ---------------------------------------------------------------- output GEMM
__global__ __launch_bounds__(256) void out_gemm_kernel(const bf16* __restrict__ ob,
                                                       const bf16* __restrict__ w,
                                                       const float* __restrict__ bo,
                                                       float* __restrict__ out) {
  __shared__ __align__(16) bf16 As[2][64][72];
  __shared__ __align__(16) bf16 Bs[2][64][72];
  int mtile = blockIdx.x, ntile = blockIdx.y;
  int tid = threadIdx.x, lane = tid & 31, wv = tid >> 5;
  int hf = lane >> 4, ln = lane & 15;
  int m0 = mtile * 64, n0 = ntile * 64;

  auto issue_stage = [&](int buf, int kc) {
#pragma unroll
    for (int c = tid; c < 512; c += 256) {
      int r = c >> 3, kk = (c & 7) * 8;
      async_ld_b128(&As[buf][r][kk], ob + (size_t)(m0 + r) * CDIM + kc + kk);
      async_ld_b128(&Bs[buf][r][kk], w  + (size_t)(n0 + r) * CDIM + kc + kk);
    }
  };

  v8f acc0 = {}; v8f acc1 = {};
  int buf = 0;
  issue_stage(0, 0);
  for (int kc = 0; kc < CDIM; kc += 64) {
    if (kc + 64 < CDIM) { issue_stage(buf ^ 1, kc + 64); wait_async_le4(); }
    else                { wait_async_0(); }
    __syncthreads();
#pragma unroll
    for (int k0 = 0; k0 < 64; k0 += 32) {
      v16bf bf = load_frag_lds_bf16(&Bs[buf][0][0], (wv & 3) * 16, k0, 72, lane);
      v16bf a0 = load_frag_lds_bf16(&As[buf][0][0], (wv >> 2) * 16, k0, 72, lane);
      v16bf a1 = load_frag_lds_bf16(&As[buf][0][0], ((wv >> 2) + 2) * 16, k0, 72, lane);
      acc0 = wmma_bf16(a0, bf, acc0);
      acc1 = wmma_bf16(a1, bf, acc1);
    }
    __syncthreads();
    buf ^= 1;
  }
  int c = n0 + (wv & 3) * 16 + ln;
  float bc = bo[c];
#pragma unroll
  for (int i = 0; i < 8; ++i) {
    int r0 = m0 + (wv >> 2) * 16 + i + hf * 8;
    out[(size_t)r0 * CDIM + c] = acc0[i] + bc;
    out[(size_t)(r0 + 32) * CDIM + c] = acc1[i] + bc;
  }
}

// ---------------------------------------------------------------- attention
// One block per (b_*H + h, qtile). 16 query rows vs all 145 keys (padded 160).
__global__ __launch_bounds__(256) void attn_kernel(const bf16* __restrict__ q,
                                                   const bf16* __restrict__ kmat,
                                                   const bf16* __restrict__ vmat,
                                                   const float* __restrict__ rpb,
                                                   const float* __restrict__ mask,
                                                   float* __restrict__ ga_buf,
                                                   bf16* __restrict__ o) {
  __shared__ __align__(16) bf16 qs[16][72];
  __shared__ __align__(16) bf16 ks[160][72];
  __shared__ __align__(16) bf16 vst[64][168];   // v transposed: [d][n]
  __shared__ __align__(16) float Ss[16][168];
  int bh = blockIdx.x, qtile = blockIdx.y;
  int b_ = bh >> 3, h = bh & 7;
  int tid = threadIdx.x, lane = tid & 31, wv = tid >> 5;
  int hf = lane >> 4, ln = lane & 15;
  size_t kb = (size_t)bh * SEQ_N * HD;
  int qrow0 = qtile * 16;

  for (int cidx = tid; cidx < 1280; cidx += 256) {
    int r = cidx >> 3, kk = (cidx & 7) * 8;
    uint4 kv; kv.x = kv.y = kv.z = kv.w = 0u;
    if (r < SEQ_N) kv = *(const uint4*)(kmat + kb + (size_t)r * HD + kk);
    *(uint4*)&ks[r][kk] = kv;
    union { uint4 u; bf16 e[8]; } tv;
    tv.u.x = tv.u.y = tv.u.z = tv.u.w = 0u;
    if (r < SEQ_N) tv.u = *(const uint4*)(vmat + kb + (size_t)r * HD + kk);
#pragma unroll
    for (int e2 = 0; e2 < 8; ++e2) vst[kk + e2][r] = tv.e[e2];
  }
  if (tid < 128) {
    int r = tid >> 3, kk = (tid & 7) * 8;
    uint4 qv; qv.x = qv.y = qv.z = qv.w = 0u;
    if (qrow0 + r < SEQ_N) qv = *(const uint4*)(q + kb + (size_t)(qrow0 + r) * HD + kk);
    *(uint4*)&qs[r][kk] = qv;
  }
  __syncthreads();

  // S = (q k^T) * scale + bias + mask   (10 column tiles over 8 waves)
  for (int pass = 0; pass < 2; ++pass) {
    if (pass == 1 && wv >= 2) break;
    int tc = wv + pass * 8;
    v8f acc = {};
#pragma unroll
    for (int k0 = 0; k0 < 64; k0 += 32) {
      v16bf a = load_frag_lds_bf16(&qs[0][0], 0, k0, 72, lane);
      v16bf b = load_frag_lds_bf16(&ks[0][0], tc * 16, k0, 72, lane);
      acc = wmma_bf16(a, b, acc);
    }
#pragma unroll
    for (int i = 0; i < 8; ++i) {
      int grow = qrow0 + i + hf * 8;
      int gcol = tc * 16 + ln;
      float sv = acc[i] * 0.125f;          // hd^-0.5
      if (gcol >= SEQ_N) {
        sv = -1e30f;                       // key padding
      } else {
        if (grow < MTOK && gcol < MTOK) {  // relative position bias (window block only)
          int ih = grow / 12, iw = grow % 12, jh = gcol / 12, jw = gcol % 12;
          int ridx = (ih - jh + 11) * 23 + (iw - jw + 11);
          sv += rpb[ridx * HEADS + h];
        }
        if (grow < SEQ_N) sv += mask[((size_t)(b_ & 3) * SEQ_N + grow) * SEQ_N + gcol];
      }
      Ss[i + hf * 8][gcol] = sv;
    }
  }
  __syncthreads();

  // global-token logit row (row 144 == local row 0 of qtile 9), pre-softmax
  if (qtile == 9 && tid < SEQ_N) ga_buf[(size_t)bh * SEQ_N + tid] = Ss[0][tid];
  __syncthreads();

  // row softmax (wave32 shuffle reductions), 2 rows per wave
  for (int rr = wv * 2; rr < wv * 2 + 2; ++rr) {
    float mx = -3.4e38f;
    for (int j = lane; j < 160; j += 32) mx = fmaxf(mx, Ss[rr][j]);
#pragma unroll
    for (int o2 = 16; o2 > 0; o2 >>= 1) mx = fmaxf(mx, __shfl_xor(mx, o2, 32));
    float sm = 0.f;
    for (int j = lane; j < 160; j += 32) { float e = __expf(Ss[rr][j] - mx); Ss[rr][j] = e; sm += e; }
#pragma unroll
    for (int o2 = 16; o2 > 0; o2 >>= 1) sm += __shfl_xor(sm, o2, 32);
    float inv = 1.0f / sm;
    for (int j = lane; j < 160; j += 32) Ss[rr][j] *= inv;
  }
  __syncthreads();

  // out = P @ V  (4 output col tiles of 16, K = 160)
  if (wv < 4) {
    v8f acc = {};
#pragma unroll
    for (int k0 = 0; k0 < 160; k0 += 32) {
      v16bf a = load_frag_lds_f32(&Ss[0][0], 0, k0, 168, lane);
      v16bf b = load_frag_lds_bf16(&vst[0][0], wv * 16, k0, 168, lane);
      acc = wmma_bf16(a, b, acc);
    }
    int gcol = h * HD + wv * 16 + ln;
#pragma unroll
    for (int i = 0; i < 8; ++i) {
      int grow = qrow0 + i + hf * 8;
      if (grow < SEQ_N) o[((size_t)b_ * SEQ_N + grow) * CDIM + gcol] = (bf16)acc[i];
    }
  }
}

// ------------------------------------------------- global-token softmax (577)
__global__ __launch_bounds__(256) void gsoftmax_kernel(const float* __restrict__ gab,
                                                       float* __restrict__ gaf) {
  int b = blockIdx.x >> 3, h = blockIdx.x & 7;
  int tid = threadIdx.x;
  __shared__ float red[256];
  float vals[3]; int nt = 0;
  float lmax = -3.4e38f;
  for (int t = tid; t < 577; t += 256) {
    float vv;
    if (t < 576) {
      int w = t / MTOK, j = t % MTOK;
      vv = gab[((size_t)((b * NWIN + w) * HEADS + h)) * SEQ_N + j];
    } else {
      vv = 0.f;
      for (int w = 0; w < NWIN; ++w)
        vv += gab[((size_t)((b * NWIN + w) * HEADS + h)) * SEQ_N + MTOK];
      vv *= 0.25f;
    }
    vals[nt++] = vv;
    lmax = fmaxf(lmax, vv);
  }
  red[tid] = lmax; __syncthreads();
  for (int s = 128; s > 0; s >>= 1) { if (tid < s) red[tid] = fmaxf(red[tid], red[tid + s]); __syncthreads(); }
  float gmax = red[0]; __syncthreads();
  float lsum = 0.f;
  for (int i = 0; i < nt; ++i) { vals[i] = __expf(vals[i] - gmax); lsum += vals[i]; }
  red[tid] = lsum; __syncthreads();
  for (int s = 128; s > 0; s >>= 1) { if (tid < s) red[tid] += red[tid + s]; __syncthreads(); }
  float inv = 1.f / red[0];
  int idx = 0;
  for (int t = tid; t < 577; t += 256) {
    float p = vals[idx++] * inv;
    if (t < 576) {
      int w = t / MTOK, j = t % MTOK;
      gaf[((size_t)((b * NWIN + w) * HEADS + h)) * SEQ_N + j] = p;
    } else {
      for (int w = 0; w < NWIN; ++w)
        gaf[((size_t)((b * NWIN + w) * HEADS + h)) * SEQ_N + MTOK] = p;
    }
  }
}

// ------------------------------------------------- gx = ga_full @ v, window-sum
__global__ __launch_bounds__(64) void gx_kernel(const float* __restrict__ gaf,
                                                const bf16* __restrict__ vmat,
                                                bf16* __restrict__ o) {
  int b = blockIdx.x >> 3, h = blockIdx.x & 7;
  int d = threadIdx.x;
  float acc = 0.f;
  for (int w = 0; w < NWIN; ++w) {
    size_t base = (size_t)((b * NWIN + w) * HEADS + h) * SEQ_N;
    for (int n = 0; n < SEQ_N; ++n)
      acc += gaf[base + n] * (float)vmat[(base + n) * HD + d];
  }
  for (int w = 0; w < NWIN; ++w)
    o[((size_t)(b * NWIN + w) * SEQ_N + MTOK) * CDIM + h * HD + d] = (bf16)acc;
}

// ---------------------------------------------------------------- launcher
extern "C" void kernel_launch(void* const* d_in, const int* in_sizes, int n_in,
                              void* d_out, int out_size, void* d_ws, size_t ws_size,
                              hipStream_t stream) {
  const float* x    = (const float*)d_in[0];
  const float* mask = (const float*)d_in[1];
  const float* rpb  = (const float*)d_in[2];
  const float* Wq   = (const float*)d_in[3];
  const float* bq   = (const float*)d_in[4];
  const float* Wk   = (const float*)d_in[5];
  const float* bk   = (const float*)d_in[6];
  const float* Wv   = (const float*)d_in[7];
  const float* bv   = (const float*)d_in[8];
  const float* Wo   = (const float*)d_in[9];
  const float* bo   = (const float*)d_in[10];

  char* ws = (char*)d_ws;
  size_t off = 0;
  auto alloc = [&](size_t bytes) -> void* {
    void* p = ws + off;
    off += (bytes + 255) & ~(size_t)255;
    return p;
  };
  const size_t WSZ = (size_t)CDIM * CDIM;               // 262144
  const size_t XSZ = (size_t)B_TOT * SEQ_N * CDIM;      // 38,010,880
  bf16* wbf  = (bf16*)alloc(4 * WSZ * sizeof(bf16));
  bf16* xbf  = (bf16*)alloc(XSZ * sizeof(bf16));        // reused as o after QKV
  bf16* qbf  = (bf16*)alloc(XSZ * sizeof(bf16));
  bf16* kbf  = (bf16*)alloc(XSZ * sizeof(bf16));
  bf16* vbf  = (bf16*)alloc(XSZ * sizeof(bf16));
  float* ga_buf  = (float*)alloc((size_t)B_TOT * HEADS * SEQ_N * sizeof(float));
  float* ga_full = (float*)alloc((size_t)B_TOT * HEADS * SEQ_N * sizeof(float));
  bf16* obf = xbf;

  cvt_f32_to_bf16<<<(int)((WSZ + 255) / 256), 256, 0, stream>>>(Wq, wbf + 0 * WSZ, (int)WSZ);
  cvt_f32_to_bf16<<<(int)((WSZ + 255) / 256), 256, 0, stream>>>(Wk, wbf + 1 * WSZ, (int)WSZ);
  cvt_f32_to_bf16<<<(int)((WSZ + 255) / 256), 256, 0, stream>>>(Wv, wbf + 2 * WSZ, (int)WSZ);
  cvt_f32_to_bf16<<<(int)((WSZ + 255) / 256), 256, 0, stream>>>(Wo, wbf + 3 * WSZ, (int)WSZ);
  cvt_f32_to_bf16<<<(int)((XSZ + 255) / 256), 256, 0, stream>>>(x, xbf, (int)XSZ);

  qkv_gemm_kernel<<<dim3(1160, 8, 3), 256, 0, stream>>>(xbf, wbf, bq, bk, bv, qbf, kbf, vbf);
  attn_kernel<<<dim3(B_TOT * HEADS, 10), 256, 0, stream>>>(qbf, kbf, vbf, rpb, mask, ga_buf, obf);
  gsoftmax_kernel<<<(B_TOT / NWIN) * HEADS, 256, 0, stream>>>(ga_buf, ga_full);
  gx_kernel<<<(B_TOT / NWIN) * HEADS, 64, 0, stream>>>(ga_full, vbf, obf);
  out_gemm_kernel<<<dim3(1160, 8), 256, 0, stream>>>(obf, wbf + 3 * WSZ, bo, (float*)d_out);

  (void)in_sizes; (void)n_in; (void)out_size; (void)ws_size;
}
